// TransformerXL_61984968016243
// MI455X (gfx1250) — matrix operand order repaired and checked by reading, not compiled, not run on hardware
//
#include <hip/hip_runtime.h>
#include <hip/hip_bf16.h>

// ---------------------------------------------------------------------------
// Transformer-XL layer for MI455X (gfx1250):
//   - all matmuls via v_wmma_f32_16x16x32_f16
//   - GEMM tile staging via TDM tensor_load_to_lds (s_wait_tensorcnt)
//   - wave indices forced scalar (readfirstlane) so WMMA always runs EXEC=all-1s
// ---------------------------------------------------------------------------

typedef _Float16 v8h  __attribute__((ext_vector_type(8)));
typedef _Float16 v16h __attribute__((ext_vector_type(16)));
typedef float    v8f  __attribute__((ext_vector_type(8)));
typedef unsigned int u32x4 __attribute__((ext_vector_type(4)));
typedef int          i32x8 __attribute__((ext_vector_type(8)));
typedef int          i32x4 __attribute__((ext_vector_type(4)));

static constexpr int kB  = 4;
static constexpr int kS  = 4096;
static constexpr int kD  = 1024;
static constexpr int kH  = 16;
static constexpr int kDK = 64;
static constexpr int kL  = 512;   // BLOCK
static constexpr int kHID = 4096;

#define CAT16(lo, hi2) __builtin_shufflevector(lo, hi2, 0,1,2,3,4,5,6,7,8,9,10,11,12,13,14,15)

// Wave id as an SGPR value -> scalar branches, EXEC stays full across WMMA.
__device__ __forceinline__ int wave_id() {
  return __builtin_amdgcn_readfirstlane((int)threadIdx.x >> 5);
}

// A-fragment (16x32 f16, ISA layout): lane half 'hi' holds K = hi*8+{0..7} and 16+hi*8+{0..7}
__device__ __forceinline__ v16h afrag_load(const _Float16* rowp, int hi) {
  v8h lo = *(const v8h*)(rowp + hi * 8);
  v8h h2 = *(const v8h*)(rowp + 16 + hi * 8);
  return CAT16(lo, h2);
}
// B-fragment (32x16 f16): lane = column n, lane half 'hi' holds K = hi*8+{0..7} and 16+hi*8+{0..7}
__device__ __forceinline__ v16h bfrag_load(const _Float16* rowp, int hi) {
  v8h lo = *(const v8h*)(rowp + hi * 8);
  v8h h2 = *(const v8h*)(rowp + 16 + hi * 8);
  return CAT16(lo, h2);
}
__device__ __forceinline__ v8f wmma16(v16h a, v16h b, v8f c) {
  return __builtin_amdgcn_wmma_f32_16x16x32_f16(false, a, false, b, (short)0, c, false, false);
}

// ---------------------------------------------------------------------------
// TDM support (CDNA5 Tensor Data Mover): DMA a 128-row x 64-byte tile into LDS
// with 16B of padding per row -> LDS row stride 80B (= 40 halves).
// ---------------------------------------------------------------------------
#if __has_builtin(__builtin_amdgcn_tensor_load_to_lds) && __has_builtin(__builtin_amdgcn_s_wait_tensorcnt)
#define USE_TDM 1
#else
#define USE_TDM 0
#endif

#if USE_TDM
typedef const void __attribute__((address_space(3)))* lds_cvptr;
__device__ __forceinline__ unsigned lds_off_of(const void* p) {
  return (unsigned)(unsigned long long)(lds_cvptr)p;   // generic -> LDS offset
}

// Tile: tile_dim0 = 8 (x 8B = 64B row), tile_dim1 = 128 rows.
// rowStrideUnits = global row stride in 8-byte units (= K/4 for f16 rows of K).
// pad: data written in 64B rows, +4 dwords pad each -> LDS stride 80B.
__device__ __forceinline__ void tdm_load_tile_128x64B(const void* gptr,
                                                      unsigned ldsByteOff,
                                                      unsigned rowStrideUnits) {
  unsigned long long ga = (unsigned long long)gptr;
  u32x4 g0;
  g0[0] = 1u;                                            // count=1 (valid user D#)
  g0[1] = ldsByteOff;                                    // lds_addr (bytes)
  g0[2] = (unsigned)(ga & 0xffffffffu);                  // global_addr[31:0]
  g0[3] = (unsigned)((ga >> 32) & 0x01ffffffu) | (2u << 30); // addr[56:32] | type=2
  i32x8 g1;
  // data_size=3 (8B) | pad_enable | pad_interval=3 (pad every 64B) | pad_amount=3 (4 dwords)
  g1[0] = (int)((3u << 16) | (1u << 20) | (3u << 22) | (3u << 25));
  g1[1] = (int)((rowStrideUnits & 0xffffu) << 16);       // tensor_dim0 lo16 (bits 63:48)
  g1[2] = (int)(((rowStrideUnits >> 16) & 0xffffu) | (128u << 16)); // dim0 hi16 | tensor_dim1 lo16
  g1[3] = (int)(8u << 16);                               // tensor_dim1 hi16=0 | tile_dim0=8
  g1[4] = (int)128u;                                     // tile_dim1=128 | tile_dim2=0
  g1[5] = (int)rowStrideUnits;                           // tensor_dim0_stride lo32
  g1[6] = 0;                                             // stride hi16 | dim1_stride lo16
  g1[7] = 0;
  i32x4 z4 = {0, 0, 0, 0};
  i32x8 z8 = {0, 0, 0, 0, 0, 0, 0, 0};
  // 6-arg form (clang-23 / therock headers): groups 0..3 + extra group + cpol
  __builtin_amdgcn_tensor_load_to_lds(g0, g1, z4, z4, z8, 0);
}
#endif

// ---------------------------------------------------------------------------
// Generic tiled GEMM: C[M][N] = A[M][K] (f16 row-major) x Bt[N][K] (f16 K-major)
// Tile 128x128, 256 threads (8 waves), wave sub-tile 64x32 (4x2 WMMA tiles).
// ---------------------------------------------------------------------------
template <bool BIAS, bool RELU, bool ADDEND, bool OUTF32>
__global__ __launch_bounds__(256) void gemm_kernel(
    const _Float16* __restrict__ A, const _Float16* __restrict__ Bt,
    const float* __restrict__ bias, const float* __restrict__ addend,
    float* __restrict__ Cf32, _Float16* __restrict__ Cf16,
    int M, int N, int K)
{
  __shared__ _Float16 As[128 * 40];
  __shared__ _Float16 Bs[128 * 40];
  const int tid = threadIdx.x, lane = tid & 31;
  const int wave = wave_id();
  const int m0 = blockIdx.x * 128, n0 = blockIdx.y * 128;
  const int wm = (wave >> 2) * 64, wn = (wave & 3) * 32;
  const int mh = lane & 15, hi = lane >> 4;
  v8f acc[4][2] = {};

  for (int k0 = 0; k0 < K; k0 += 32) {
#if USE_TDM
    if (wave == 0) {
      tdm_load_tile_128x64B(&A [(size_t)m0 * K + k0], lds_off_of(As), (unsigned)(K >> 2));
      tdm_load_tile_128x64B(&Bt[(size_t)n0 * K + k0], lds_off_of(Bs), (unsigned)(K >> 2));
      __builtin_amdgcn_s_wait_tensorcnt(0);
    }
#else
#pragma unroll
    for (int r = 0; r < 2; ++r) {
      int lin = tid * 8 + r * 2048;
      int row = lin >> 5, col = lin & 31;
      *(v8h*)&As[row * 40 + col] = *(const v8h*)&A [(size_t)(m0 + row) * K + k0 + col];
      *(v8h*)&Bs[row * 40 + col] = *(const v8h*)&Bt[(size_t)(n0 + row) * K + k0 + col];
    }
#endif
    __syncthreads();
    v16h af[4], bf2[2];
#pragma unroll
    for (int i = 0; i < 4; ++i) af[i]  = afrag_load(&As[(wm + i * 16 + mh) * 40], hi);
#pragma unroll
    for (int j = 0; j < 2; ++j) bf2[j] = bfrag_load(&Bs[(wn + j * 16 + mh) * 40], hi);
#pragma unroll
    for (int i = 0; i < 4; ++i)
#pragma unroll
      for (int j = 0; j < 2; ++j)
        acc[i][j] = wmma16(af[i], bf2[j], acc[i][j]);
    __syncthreads();
  }

#pragma unroll
  for (int i = 0; i < 4; ++i)
#pragma unroll
    for (int j = 0; j < 2; ++j) {
      int col = n0 + wn + j * 16 + mh;
      float bi = BIAS ? bias[col] : 0.f;
#pragma unroll
      for (int e = 0; e < 8; ++e) {
        int row = m0 + wm + i * 16 + hi * 8 + e;
        float v = acc[i][j][e] + bi;
        if (RELU) v = v > 0.f ? v : 0.f;
        if (ADDEND) v += addend[(size_t)row * N + col];
        if (OUTF32) Cf32[(size_t)row * N + col] = v;
        else        Cf16[(size_t)row * N + col] = (_Float16)v;
      }
    }
}

// ---------------------------------------------------------------------------
// Weight convert+transpose: W[K][N] f32 -> out[N][K] f16
// ---------------------------------------------------------------------------
__global__ __launch_bounds__(256) void wtrans_kernel(const float* __restrict__ W,
                                                     _Float16* __restrict__ out, int K, int N) {
  int idx = blockIdx.x * 256 + threadIdx.x;
  if (idx >= K * N) return;
  int k = idx / N, n = idx % N;
  out[(size_t)n * K + k] = (_Float16)W[idx];
}

// ---------------------------------------------------------------------------
// Relative positional encoding P[T][1024] f16: [sin(ang) | cos(ang)]
// ---------------------------------------------------------------------------
__global__ __launch_bounds__(256) void relpos_kernel(_Float16* __restrict__ P, int T) {
  int idx = blockIdx.x * 256 + threadIdx.x;
  if (idx >= T * kD) return;
  int t = idx >> 10, d = idx & 1023;
  float pos = (float)(T - 1 - t);
  int j = d < 512 ? d : d - 512;
  float invf = __expf(-(float)(2 * j) * (logf(10000.f) / 1024.f));
  float a = pos * invf;
  P[idx] = (_Float16)(d < 512 ? __sinf(a) : __cosf(a));
}

// ---------------------------------------------------------------------------
// LayerNorm over D=1024: out_f16 = LN(x (+ res)) * g + b. One WG per row.
// ---------------------------------------------------------------------------
__global__ __launch_bounds__(256) void layernorm_kernel(
    const float* __restrict__ x, const float* __restrict__ res,
    const float* __restrict__ g, const float* __restrict__ bta,
    _Float16* __restrict__ out)
{
  __shared__ float red[256];
  __shared__ float red2[256];
  size_t base = (size_t)blockIdx.x * kD;
  int tid = threadIdx.x;
  float s = 0.f, s2 = 0.f;
  for (int i = tid; i < kD; i += 256) {
    float v = x[base + i] + (res ? res[base + i] : 0.f);
    s += v; s2 += v * v;
  }
  red[tid] = s; red2[tid] = s2; __syncthreads();
  for (int st = 128; st > 0; st >>= 1) {
    if (tid < st) { red[tid] += red[tid + st]; red2[tid] += red2[tid + st]; }
    __syncthreads();
  }
  float mean = red[0] * (1.f / kD);
  float var  = red2[0] * (1.f / kD) - mean * mean;
  float inv  = rsqrtf(var + 1e-5f);
  for (int i = tid; i < kD; i += 256) {
    float v = x[base + i] + (res ? res[base + i] : 0.f);
    out[base + i] = (_Float16)((v - mean) * inv * g[i] + bta[i]);
  }
}

// ---------------------------------------------------------------------------
// Attention scores: scores[b,h,l,t] = scale*(ac + rel_shift(bd))
//   rel_shift(bd)[l,t] = bd[l, c], c = t + L-1 - l  (0 if c >= T)
// bd for a 64l x 64t chunk needs <=144 c-columns -> WMMA into LDS, then gather.
// All job loops are scalar (wave id in SGPR) so WMMA runs with EXEC=all-1s.
// ---------------------------------------------------------------------------
__global__ __launch_bounds__(256) void attn_scores_kernel(
    const _Float16* __restrict__ Qu, const _Float16* __restrict__ Qv,
    const _Float16* __restrict__ Kb, const _Float16* __restrict__ R,
    float* __restrict__ scores, int blk, int T, int ctx0)
{
  __shared__ float bdbuf[64 * 160];
  const int tid = threadIdx.x, lane = tid & 31;
  const int wave = wave_id();
  const int lt = blockIdx.x & 7;
  const int h  = (blockIdx.x >> 3) & 15;
  const int b  = blockIdx.x >> 7;
  const int l0 = lt * 64;
  const int mh = lane & 15, hi = lane >> 4;
  const float scale = 0.125f;  // 1/sqrt(64)

  for (int t0 = 0; t0 < T; t0 += 64) {
    int cmin = t0 + kL - 64 - l0;          // >= 0 since l0 <= L-64
    int c0   = cmin & ~15;
    int cend = t0 + 63 + (kL - 1) - l0 + 1;
    if (cend > T) cend = T;
    int nct = (cend - c0 + 15) >> 4;       // <= 9

    // ---- BD phase ----
    for (int job = wave; job < 4 * nct; job += 8) {
      int mi = job & 3, ni = job >> 2;
      const _Float16* arow =
          Qv + ((size_t)(b * kS + blk * kL + l0 + mi * 16 + mh)) * kD + h * kDK;
      int c = c0 + ni * 16 + mh;
      int cl = c < T ? c : (T - 1);
      const _Float16* brow = R + (size_t)cl * kD + h * kDK;
      v8f acc = {};
#pragma unroll
      for (int ks = 0; ks < 2; ++ks)
        acc = wmma16(afrag_load(arow + ks * 32, hi), bfrag_load(brow + ks * 32, hi), acc);
      int ccol = ni * 16 + mh;
#pragma unroll
      for (int e = 0; e < 8; ++e)
        bdbuf[(mi * 16 + hi * 8 + e) * 160 + ccol] = acc[e];
    }
    __syncthreads();

    // ---- AC phase + shifted-BD gather + store ----
    for (int job = wave; job < 16; job += 8) {
      int mi = job & 3, ni = job >> 2;
      const _Float16* arow =
          Qu + ((size_t)(b * kS + blk * kL + l0 + mi * 16 + mh)) * kD + h * kDK;
      int trow = t0 + ni * 16 + mh;
      const _Float16* brow = Kb + (size_t)(b * kS + ctx0 + trow) * kD + h * kDK;
      v8f acc = {};
#pragma unroll
      for (int ks = 0; ks < 2; ++ks)
        acc = wmma16(afrag_load(arow + ks * 32, hi), bfrag_load(brow + ks * 32, hi), acc);
#pragma unroll
      for (int e = 0; e < 8; ++e) {
        int l = l0 + mi * 16 + hi * 8 + e;
        int t = t0 + ni * 16 + mh;
        int c = t + (kL - 1) - l;
        float v = acc[e];
        if (c < T) v += bdbuf[(l - l0) * 160 + (c - c0)];
        scores[(((size_t)(b * kH + h)) * kL + l) * 1024 + t] = v * scale;
      }
    }
    __syncthreads();
  }
}

// ---------------------------------------------------------------------------
// Row softmax over T, write f16 probs. Grid: B*H*L rows.
// ---------------------------------------------------------------------------
__global__ __launch_bounds__(256) void softmax_kernel(const float* __restrict__ scores,
                                                      _Float16* __restrict__ probs, int T) {
  __shared__ float red[256];
  size_t base = (size_t)blockIdx.x * 1024;
  int tid = threadIdx.x;
  float m = -1e30f;
  for (int t = tid; t < T; t += 256) m = fmaxf(m, scores[base + t]);
  red[tid] = m; __syncthreads();
  for (int s = 128; s > 0; s >>= 1) { if (tid < s) red[tid] = fmaxf(red[tid], red[tid + s]); __syncthreads(); }
  m = red[0]; __syncthreads();
  float sum = 0.f;
  for (int t = tid; t < T; t += 256) sum += __expf(scores[base + t] - m);
  red[tid] = sum; __syncthreads();
  for (int s = 128; s > 0; s >>= 1) { if (tid < s) red[tid] += red[tid + s]; __syncthreads(); }
  float inv = 1.f / red[0];
  for (int t = tid; t < T; t += 256)
    probs[base + t] = (_Float16)(__expf(scores[base + t] - m) * inv);
}

// ---------------------------------------------------------------------------
// PV: O[b, blk*L+l, h*64+dk] = sum_t probs[b,h,l,t] * V[b, ctx0+t, h, dk]
// V transposed into LDS in K-chunks of 256; 16 WMMA tile jobs (2/wave).
// ---------------------------------------------------------------------------
__global__ __launch_bounds__(256) void attn_pv_kernel(
    const _Float16* __restrict__ probs, const _Float16* __restrict__ Vb,
    _Float16* __restrict__ Obuf, int blk, int T, int ctx0)
{
  __shared__ _Float16 Vt[64 * 264];   // [dk][t-chunk], stride 264 halves
  const int tid = threadIdx.x, lane = tid & 31;
  const int wave = wave_id();
  const int lt = blockIdx.x & 7;
  const int h  = (blockIdx.x >> 3) & 15;
  const int b  = blockIdx.x >> 7;
  const int l0 = lt * 64;
  const int mh = lane & 15, hi = lane >> 4;
  const int mi0 = wave & 3,       ni0 = wave >> 2;
  const int mi1 = (wave + 8) & 3, ni1 = (wave + 8) >> 2;
  v8f acc0 = {}, acc1 = {};

  for (int kc = 0; kc < T; kc += 256) {
    for (int idx = tid; idx < 256 * 8; idx += 256) {
      int t = idx >> 3, dq = (idx & 7) * 8;
      const _Float16* src = &Vb[(size_t)(b * kS + ctx0 + kc + t) * kD + h * kDK + dq];
      v8h val = *(const v8h*)src;
      if (kc + 256 < T)
        __builtin_prefetch(src + (size_t)256 * kD, 0, 0);   // global_prefetch_b8
#pragma unroll
      for (int j = 0; j < 8; ++j) Vt[(dq + j) * 264 + t] = val[j];
    }
    __syncthreads();
    for (int ks = 0; ks < 256; ks += 32) {
      {
        const _Float16* arow =
            probs + ((size_t)((b * kH + h) * kL + l0 + mi0 * 16 + mh)) * 1024 + kc + ks;
        acc0 = wmma16(afrag_load(arow, hi),
                      bfrag_load(&Vt[(ni0 * 16 + mh) * 264 + ks], hi), acc0);
      }
      {
        const _Float16* arow =
            probs + ((size_t)((b * kH + h) * kL + l0 + mi1 * 16 + mh)) * 1024 + kc + ks;
        acc1 = wmma16(afrag_load(arow, hi),
                      bfrag_load(&Vt[(ni1 * 16 + mh) * 264 + ks], hi), acc1);
      }
    }
    __syncthreads();
  }
#pragma unroll
  for (int e = 0; e < 8; ++e) {
    int l = l0 + mi0 * 16 + hi * 8 + e, dk = ni0 * 16 + mh;
    Obuf[(size_t)(b * kS + blk * kL + l) * kD + h * kDK + dk] = (_Float16)acc0[e];
  }
#pragma unroll
  for (int e = 0; e < 8; ++e) {
    int l = l0 + mi1 * 16 + hi * 8 + e, dk = ni1 * 16 + mh;
    Obuf[(size_t)(b * kS + blk * kL + l) * kD + h * kDK + dk] = (_Float16)acc1[e];
  }
}

// ---------------------------------------------------------------------------
extern "C" void kernel_launch(void* const* d_in, const int* in_sizes, int n_in,
                              void* d_out, int out_size, void* d_ws, size_t ws_size,
                              hipStream_t stream) {
  const float* x  = (const float*)d_in[0];
  const float* g1 = (const float*)d_in[1];
  const float* b1 = (const float*)d_in[2];
  const float* g2 = (const float*)d_in[3];
  const float* b2 = (const float*)d_in[4];
  const float* Wq = (const float*)d_in[5];
  const float* Wk = (const float*)d_in[6];
  const float* Wv = (const float*)d_in[7];
  const float* Wr = (const float*)d_in[8];
  const float* u  = (const float*)d_in[9];
  const float* v  = (const float*)d_in[10];
  const float* Wo = (const float*)d_in[11];
  const float* bo = (const float*)d_in[12];
  const float* Wh = (const float*)d_in[13];
  const float* bh = (const float*)d_in[14];
  const float* Wf = (const float*)d_in[15];
  const float* bf = (const float*)d_in[16];

  char* wsb = (char*)d_ws;
  size_t off = 0;
  auto alloc = [&](size_t bytes) -> void* {
    off = (off + 255) & ~(size_t)255;
    void* p = wsb + off;
    off += bytes;
    return p;
  };

  const size_t MT = (size_t)kB * kS;  // 16384 rows
  _Float16* WqT = (_Float16*)alloc((size_t)kD * kD * 2);
  _Float16* WkT = (_Float16*)alloc((size_t)kD * kD * 2);
  _Float16* WvT = (_Float16*)alloc((size_t)kD * kD * 2);
  _Float16* WrT = (_Float16*)alloc((size_t)kD * kD * 2);
  _Float16* WoT = (_Float16*)alloc((size_t)kD * kD * 2);
  _Float16* WhT = (_Float16*)alloc((size_t)kD * kHID * 2);
  _Float16* WfT = (_Float16*)alloc((size_t)kHID * kD * 2);
  _Float16* ln   = (_Float16*)alloc(MT * kD * 2);
  _Float16* Qu   = (_Float16*)alloc(MT * kD * 2);
  _Float16* Qv   = (_Float16*)alloc(MT * kD * 2);
  _Float16* Kb   = (_Float16*)alloc(MT * kD * 2);
  _Float16* Vb   = (_Float16*)alloc(MT * kD * 2);
  _Float16* P512  = (_Float16*)alloc((size_t)512 * kD * 2);
  _Float16* P1024 = (_Float16*)alloc((size_t)1024 * kD * 2);
  _Float16* R512  = (_Float16*)alloc((size_t)512 * kD * 2);
  _Float16* R1024 = (_Float16*)alloc((size_t)1024 * kD * 2);
  float*    scores = (float*)   alloc((size_t)kB * kH * kL * 1024 * 4);
  _Float16* probs  = (_Float16*)alloc((size_t)kB * kH * kL * 1024 * 2);
  _Float16* Obuf   = (_Float16*)alloc(MT * kD * 2);
  float*    att    = (float*)   alloc(MT * kD * 4);
  _Float16* h1     = (_Float16*)alloc(MT * kHID * 2);
  (void)ws_size; (void)in_sizes; (void)n_in; (void)out_size;

  // 1) weight transposes (f32 -> f16 [N][K])
  wtrans_kernel<<<(kD * kD) / 256, 256, 0, stream>>>(Wq, WqT, kD, kD);
  wtrans_kernel<<<(kD * kD) / 256, 256, 0, stream>>>(Wk, WkT, kD, kD);
  wtrans_kernel<<<(kD * kD) / 256, 256, 0, stream>>>(Wv, WvT, kD, kD);
  wtrans_kernel<<<(kD * kD) / 256, 256, 0, stream>>>(Wr, WrT, kD, kD);
  wtrans_kernel<<<(kD * kD) / 256, 256, 0, stream>>>(Wo, WoT, kD, kD);
  wtrans_kernel<<<(kD * kHID) / 256, 256, 0, stream>>>(Wh, WhT, kD, kHID);
  wtrans_kernel<<<(kHID * kD) / 256, 256, 0, stream>>>(Wf, WfT, kHID, kD);

  // 2) relative positional encodings
  relpos_kernel<<<(512 * kD) / 256, 256, 0, stream>>>(P512, 512);
  relpos_kernel<<<(1024 * kD) / 256, 256, 0, stream>>>(P1024, 1024);

  // 3) LN1
  layernorm_kernel<<<(int)MT, 256, 0, stream>>>(x, nullptr, g1, b1, ln);

  // 4) projections (bias u/v folded into Qu/Qv)
  dim3 gMain(MT / 128, kD / 128);
  gemm_kernel<true,  false, false, false><<<gMain, 256, 0, stream>>>(ln, WqT, u,  nullptr, nullptr, Qu, (int)MT, kD, kD);
  gemm_kernel<true,  false, false, false><<<gMain, 256, 0, stream>>>(ln, WqT, v,  nullptr, nullptr, Qv, (int)MT, kD, kD);
  gemm_kernel<false, false, false, false><<<gMain, 256, 0, stream>>>(ln, WkT, nullptr, nullptr, nullptr, Kb, (int)MT, kD, kD);
  gemm_kernel<false, false, false, false><<<gMain, 256, 0, stream>>>(ln, WvT, nullptr, nullptr, nullptr, Vb, (int)MT, kD, kD);
  gemm_kernel<false, false, false, false><<<dim3(512 / 128, kD / 128), 256, 0, stream>>>(P512, WrT, nullptr, nullptr, nullptr, R512, 512, kD, kD);
  gemm_kernel<false, false, false, false><<<dim3(1024 / 128, kD / 128), 256, 0, stream>>>(P1024, WrT, nullptr, nullptr, nullptr, R1024, 1024, kD, kD);

  // 5) blockwise attention (blocks independent; scratch reused sequentially)
  for (int blk = 0; blk < 8; ++blk) {
    int T = (blk == 0) ? 512 : 1024;
    int ctx0 = (blk == 0) ? 0 : (blk - 1) * kL;
    const _Float16* R = (blk == 0) ? R512 : R1024;
    attn_scores_kernel<<<kB * kH * (kL / 64), 256, 0, stream>>>(Qu, Qv, Kb, R, scores, blk, T, ctx0);
    softmax_kernel<<<kB * kH * kL, 256, 0, stream>>>(scores, probs, T);
    attn_pv_kernel<<<kB * kH * (kL / 64), 256, 0, stream>>>(probs, Vb, Obuf, blk, T, ctx0);
  }

  // 6) output projection -> att (f32)
  gemm_kernel<true, false, false, true><<<gMain, 256, 0, stream>>>(Obuf, WoT, bo, nullptr, att, nullptr, (int)MT, kD, kD);

  // 7) LN2 over (att + x) -> ln (f16, reused)
  layernorm_kernel<<<(int)MT, 256, 0, stream>>>(att, x, g2, b2, ln);

  // 8) FFN
  gemm_kernel<true, true,  false, false><<<dim3(MT / 128, kHID / 128), 256, 0, stream>>>(ln, WhT, bh, nullptr, nullptr, h1, (int)MT, kHID, kD);
  gemm_kernel<true, false, true,  true ><<<gMain, 256, 0, stream>>>(h1, WfT, bf, att, (float*)d_out, nullptr, (int)MT, kD, kHID);
}
